// Masked_Linear_74680891343304
// MI455X (gfx1250) — matrix operand order repaired and checked by reading, not compiled, not run on hardware
//
#include <hip/hip_runtime.h>

// Block-diagonal masked linear for MI455X (gfx1250, wave32, WMMA + TDM).
// y[b, p*820+o] = sum_k x[b, p*8192+k] * W[p*820+o, p*8192+k] + bias[p*820+o]

namespace {
constexpr int PIECES = 5;
constexpr int OUT_F  = 4100;
constexpr int IN_F   = 40960;
constexpr int BO     = OUT_F / PIECES;  // 820
constexpr int BI     = IN_F / PIECES;   // 8192
constexpr int BATCH  = 256;

constexpr int WG_M   = 128;             // 8 waves x 16 rows
constexpr int WAVE_N = 64;              // 4 x 16-col WMMA tiles per wave
constexpr int KSTEP  = 32;              // K per WMMA step
constexpr int F_PAD  = 36;              // f32 LDS row stride (32 + 4 pad dwords, TDM pad_enable)
constexpr int B_PAD  = 40;              // bf16 LDS row stride (bank-conflict-free frag reads)
}

typedef __attribute__((ext_vector_type(16))) __bf16 v16bf;
typedef __attribute__((ext_vector_type(8)))  __bf16 v8bf;
typedef __attribute__((ext_vector_type(8)))  float  v8f;
typedef __attribute__((ext_vector_type(4)))  float  f32x4;
typedef __attribute__((ext_vector_type(4)))  unsigned int u32x4;
typedef __attribute__((ext_vector_type(8)))  int    i32x8;
typedef __attribute__((ext_vector_type(4)))  int    i32x4;

static __device__ __forceinline__ v16bf cvt16(f32x4 v0, f32x4 v1, f32x4 v2, f32x4 v3) {
  v16bf r;
  r[0]  = (__bf16)v0.x;  r[1]  = (__bf16)v0.y;  r[2]  = (__bf16)v0.z;  r[3]  = (__bf16)v0.w;
  r[4]  = (__bf16)v1.x;  r[5]  = (__bf16)v1.y;  r[6]  = (__bf16)v1.z;  r[7]  = (__bf16)v1.w;
  r[8]  = (__bf16)v2.x;  r[9]  = (__bf16)v2.y;  r[10] = (__bf16)v2.z;  r[11] = (__bf16)v2.w;
  r[12] = (__bf16)v3.x;  r[13] = (__bf16)v3.y;  r[14] = (__bf16)v3.z;  r[15] = (__bf16)v3.w;
  return r;
}

// Loop-invariant TDM descriptor group 1: 2D tensor [tensor_d1 rows x BI cols], f32,
// tile = [64 rows x 32 K], LDS rows padded to 36 dwords via pad_enable.
static __device__ __forceinline__ i32x8 tdm_make_g1(unsigned tensor_d1) {
  i32x8 g1;
  g1[0] = (int)((2u << 16)        // data_size = 4 bytes
              | (1u << 20)        // pad_enable
              | (4u << 22)        // pad_interval: 32 dwords
              | (3u << 25));      // pad_amount: 4 dwords -> 36-dword LDS rows
  g1[1] = (int)(((unsigned)BI & 0xFFFFu) << 16);                        // tensor_dim0[15:0]
  g1[2] = (int)(((unsigned)BI >> 16) | ((tensor_d1 & 0xFFFFu) << 16));  // dim0[31:16], dim1[15:0]
  g1[3] = (int)((tensor_d1 >> 16) | ((unsigned)KSTEP << 16));           // dim1[31:16], tile_dim0=32
  g1[4] = (int)(unsigned)WAVE_N;                                        // tile_dim1=64, tile_dim2=0
  g1[5] = (int)(unsigned)IN_F;                                          // tensor_dim0_stride
  g1[6] = 0;
  g1[7] = 0;
  return g1;
}

// Issue the DMA: only group 0 (lds_addr + global_addr) varies per iteration.
static __device__ __forceinline__ void tdm_issue(unsigned lds_off, const float* gtile,
                                                 i32x8 g1) {
  unsigned long long ga = (unsigned long long)(uintptr_t)gtile;
  u32x4 g0;
  g0[0] = 1u;                                         // count=1 (valid user D#)
  g0[1] = lds_off;                                    // lds_addr
  g0[2] = (unsigned)ga;                               // global_addr[31:0]
  g0[3] = (unsigned)((ga >> 32) & 0x01FFFFFFu)        // global_addr[56:32]
        | (2u << 30);                                 // type=2 ("image")
  i32x4 z4 = {0, 0, 0, 0};
#if defined(__clang_major__) && (__clang_major__ >= 23)
  i32x8 z8 = {0, 0, 0, 0, 0, 0, 0, 0};
  __builtin_amdgcn_tensor_load_to_lds(g0, g1, z4, z4, z8, 0);
#else
  __builtin_amdgcn_tensor_load_to_lds(g0, g1, z4, z4, 0);
#endif
}

__global__ __launch_bounds__(256)
void masked_linear_bdiag_wmma(const float* __restrict__ X,
                              const float* __restrict__ W,
                              const float* __restrict__ Bias,
                              float* __restrict__ Y) {
  __shared__ float  fbuf[2][WAVE_N][F_PAD];   // raw f32 weight tiles (TDM dest)
  __shared__ __bf16 bbuf[2][WAVE_N][B_PAD];   // converted bf16 weight tiles

  const int piece = blockIdx.z;
  const int nBase = blockIdx.x * WAVE_N;      // column offset within this piece's BO
  const int tid   = threadIdx.x;
  const int wave  = tid >> 5;                 // 0..7 -> M sub-tile
  const int lane  = tid & 31;
  const int lhalf = lane >> 4;
  const int l15   = lane & 15;
  const int mBase = blockIdx.y * WG_M + wave * 16;

  const float* __restrict__ Xp = X + (size_t)piece * BI;
  const float* __restrict__ Wtile =
      W + (size_t)(piece * BO + nBase) * IN_F + (size_t)piece * BI;

  // A fragment source: lanes l and l+16 share batch row M = mBase + l15.
  const float* __restrict__ xrow = Xp + (size_t)(mBase + l15) * IN_F;

  // Rows available from the tile start; OOB rows of the partial tile read as zero.
  const i32x8 g1 = tdm_make_g1((unsigned)(BO - nBase));
  const unsigned fb_off[2] = {(unsigned)(size_t)&fbuf[0][0][0],
                              (unsigned)(size_t)&fbuf[1][0][0]};

  // Cooperative convert assignment: thread -> (row, 8-elem K chunk).
  const int cn = tid >> 2;                    // 0..63
  const int ck = (tid & 3) * 8;               // 0,8,16,24

  v8f acc[4] = {};

  // Prologue: wave 0 kicks off the first weight-tile DMA.
  if (wave == 0) tdm_issue(fb_off[0], Wtile, g1);

  int pf = 0;
  for (int k = 0; k < BI; k += KSTEP) {
    // Scalar wait; no-op for waves 1..7 (their TENSORcnt is always 0).
    __builtin_amdgcn_s_wait_tensorcnt(0);
    __syncthreads();                          // fbuf[pf] visible to the whole WG

    // Prefetch next weight tile via TDM while we convert/compute this one.
    if (wave == 0 && (k + KSTEP) < BI)
      tdm_issue(fb_off[pf ^ 1], Wtile + (k + KSTEP), g1);

    // Convert staged f32 tile -> bf16 tile, once per workgroup (not per wave).
    {
      const float* fp = &fbuf[pf][cn][ck];
      f32x4 f0 = *(const f32x4*)fp;
      f32x4 f1 = *(const f32x4*)(fp + 4);
      v8bf h;
      h[0] = (__bf16)f0.x; h[1] = (__bf16)f0.y; h[2] = (__bf16)f0.z; h[3] = (__bf16)f0.w;
      h[4] = (__bf16)f1.x; h[5] = (__bf16)f1.y; h[6] = (__bf16)f1.z; h[7] = (__bf16)f1.w;
      *(v8bf*)&bbuf[pf][cn][ck] = h;
    }
    __syncthreads();                          // bbuf[pf] ready

    // ---- A fragment (16x32 bf16): lane<16 -> K [k..k+7],[k+16..k+23]; lane>=16 -> +8.
    const float* ax = xrow + k + lhalf * 8;
    f32x4 a0 = *(const f32x4*)(ax);
    f32x4 a1 = *(const f32x4*)(ax + 4);
    f32x4 a2 = *(const f32x4*)(ax + 16);
    f32x4 a3 = *(const f32x4*)(ax + 20);
    v16bf afrag = cvt16(a0, a1, a2, a3);

    if (k + 256 < BI)
      __builtin_prefetch(xrow + k + 256, 0, 0);           // global_prefetch_b8

    // ---- 4 B fragments from LDS: col = j*16+l15, K [lhalf*8..+7],[16+lhalf*8..+7].
#pragma unroll
    for (int j = 0; j < 4; ++j) {
      const __bf16* bp = &bbuf[pf][j * 16 + l15][lhalf * 8];
      v8bf lo = *(const v8bf*)bp;
      v8bf hi = *(const v8bf*)(bp + 16);
      v16bf bfrag = __builtin_shufflevector(lo, hi, 0, 1, 2, 3, 4, 5, 6, 7,
                                            8, 9, 10, 11, 12, 13, 14, 15);
      acc[j] = __builtin_amdgcn_wmma_f32_16x16x32_bf16(
          /*neg_a=*/false, afrag, /*neg_b=*/false, bfrag,
          /*c_mod=*/(short)0, acc[j], /*reuse_a=*/false, /*reuse_b=*/false);
    }
    pf ^= 1;
  }

  // ---- Epilogue: C layout -> lane col = l15, VGPR r -> row r + 8*lhalf. Add bias, store.
#pragma unroll
  for (int j = 0; j < 4; ++j) {
    const int ncol = nBase + j * 16 + l15;
    if (ncol < BO) {
      const int gcol = piece * BO + ncol;
      const float bv = Bias[gcol];
#pragma unroll
      for (int r = 0; r < 8; ++r) {
        const int m = mBase + r + lhalf * 8;
        Y[(size_t)m * OUT_F + gcol] = acc[j][r] + bv;
      }
    }
  }
}

extern "C" void kernel_launch(void* const* d_in, const int* in_sizes, int n_in,
                              void* d_out, int out_size, void* d_ws, size_t ws_size,
                              hipStream_t stream) {
  (void)in_sizes; (void)n_in; (void)d_ws; (void)ws_size; (void)out_size;
  const float* x    = (const float*)d_in[0];
  const float* w    = (const float*)d_in[1];
  const float* bias = (const float*)d_in[2];
  float* y          = (float*)d_out;

  dim3 grid((BO + WAVE_N - 1) / WAVE_N,   // 13 N-tiles
            BATCH / WG_M,                 // 2 M-tiles
            PIECES);                      // 5 pieces
  masked_linear_bdiag_wmma<<<grid, 256, 0, stream>>>(x, w, bias, y);
}
